// CCN_26998164422687
// MI455X (gfx1250) — matrix operand order repaired and checked by reading, not compiled, not run on hardware
//
#include <hip/hip_runtime.h>
#include <hip/hip_bf16.h>

// ---------------------------------------------------------------------------
// CDNA5 (gfx1250) spiking-net forward.
//   GEMMs (M=T*N=2048 rows) use v_wmma_f32_16x16x32_bf16 with LDS-staged
//   tiles, register-prefetch pipelining, and compile-time N/K so the
//   epilogue stores use immediate offsets.
//   Time scans are per-(batch,feature) register recurrences.
// ---------------------------------------------------------------------------

typedef __bf16 v16bf __attribute__((ext_vector_type(16)));
typedef __bf16 v8bf  __attribute__((ext_vector_type(8)));
typedef float  v8f   __attribute__((ext_vector_type(8)));

__device__ __forceinline__ v16bf cat16(v8bf lo, v8bf hi) {
    return __builtin_shufflevector(lo, hi, 0,1,2,3,4,5,6,7,8,9,10,11,12,13,14,15);
}

// ---------------------------------------------------------------------------
// fp32 -> bf16 weight conversion (grid-stride)
// ---------------------------------------------------------------------------
__global__ void f2bf_kernel(const float* __restrict__ w, __bf16* __restrict__ o, int n) {
    int i = blockIdx.x * blockDim.x + threadIdx.x;
    int stride = gridDim.x * blockDim.x;
    for (; i < n; i += stride) o[i] = (__bf16)w[i];
}

// ---------------------------------------------------------------------------
// Layer 0 fused: filter(0.5) -> 2x32 Jeffress mix -> LIF(1.5) -> filter(sf0)
//                -> w_cc reduce over A=32 (wave butterfly) -> IF -> spikes s0
// One wave32 per (n,c); lane index == Jeffress unit a.
// ---------------------------------------------------------------------------
__global__ __launch_bounds__(256) void l0_kernel(
    const float* __restrict__ x,      // (T,N,2,C)
    const float* __restrict__ w_jeff, // (A,2)
    const float* __restrict__ w_cc,   // (1,A)
    const float* __restrict__ w_sf0,  // scalar
    float* __restrict__ s0,           // (T,N,C) spikes out
    int T, int Nb, int C)
{
    const int gtid = blockIdx.x * blockDim.x + threadIdx.x;
    const int a   = gtid & 31;
    const int wid = gtid >> 5;            // one wave per (n,c)
    const int n = wid / C;
    const int c = wid % C;

    const float wj0 = w_jeff[a * 2 + 0];
    const float wj1 = w_jeff[a * 2 + 1];
    const float wcc = w_cc[a];
    const float sig  = 1.f / (1.f + __expf(-w_sf0[0]));
    const float dec0 = 1.f - sig;
    const float inv_lif = 1.f / 1.5f;

    float h0 = 0.f, h1 = 0.f, vl = 0.f, z = 0.f, vif = 0.f;
    for (int t = 0; t < T; ++t) {
        const size_t base = ((size_t)(t * Nb + n) * 2) * C + c;
        const float x0 = x[base];
        const float x1 = x[base + C];
        h0 = h0 * 0.5f + x0;                       // synapse_filter tau=2
        h1 = h1 * 0.5f + x1;
        const float y = h0 * wj0 + h1 * wj1;       // 2 -> A mix
        vl = vl + (y - vl) * inv_lif;              // LIF, decay_input
        const float s = (vl >= 1.f) ? 1.f : 0.f;
        vl *= (1.f - s);                           // hard reset
        z = z * dec0 + s;                          // learnable synapse filter
        float p = z * wcc;                         // w_cc reduce over a
        p += __shfl_xor(p, 16, 32);
        p += __shfl_xor(p, 8, 32);
        p += __shfl_xor(p, 4, 32);
        p += __shfl_xor(p, 2, 32);
        p += __shfl_xor(p, 1, 32);
        vif += p;                                  // IFNode
        const float sif = (vif >= 1.f) ? 1.f : 0.f;
        vif *= (1.f - sif);
        if (a == 0) s0[(size_t)(t * Nb + n) * C + c] = sif;
    }
}

// ---------------------------------------------------------------------------
// SynapseFilter scan over T; emits bf16 activations [T*N][F] (WMMA A matrix).
// ---------------------------------------------------------------------------
__global__ void filter_kernel(const float* __restrict__ s, const float* __restrict__ w_sf,
                              __bf16* __restrict__ h, int T, int NC)
{
    int i = blockIdx.x * blockDim.x + threadIdx.x;
    if (i >= NC) return;
    const float dec = 1.f - 1.f / (1.f + __expf(-w_sf[0]));
    float y = 0.f;
    for (int t = 0; t < T; ++t) {
        y = y * dec + s[(size_t)t * NC + i];
        h[(size_t)t * NC + i] = (__bf16)y;
    }
}

// ---------------------------------------------------------------------------
// IFNode scan over T, in place: reads currents, writes spikes to same buffer.
// ---------------------------------------------------------------------------
__global__ void if_kernel(float* __restrict__ cur, int T, int NC)
{
    int i = blockIdx.x * blockDim.x + threadIdx.x;
    if (i >= NC) return;
    float v = 0.f;
    for (int t = 0; t < T; ++t) {
        v += cur[(size_t)t * NC + i];
        const float s = (v >= 1.f) ? 1.f : 0.f;
        v *= (1.f - s);
        cur[(size_t)t * NC + i] = s;
    }
}

// ---------------------------------------------------------------------------
// bf16 WMMA GEMM: C[M][NN] = A[M][KK] * B[NN][KK]^T   (B = weight rows)
// NN, KK compile-time: immediate-offset epilogue stores, immediate strides.
// Workgroup: 256 threads = 8 waves; tile BM=128, BN=128, BK=32.
// Wave (wm in 0..1, wn in 0..3) owns a 64x32 sub-tile = 4x2 WMMA 16x16 frags.
// Register-prefetch: next K-tile's global loads overlap current tile's WMMAs.
// M, NN multiples of 128; KK multiple of 32 (all true for this net).
// ---------------------------------------------------------------------------
#define BM 128
#define BN 128
#define BK 32
#define LDT 40   // padded LDS row stride (bf16 elems) to spread banks

template <int NN, int KK>
__global__ __launch_bounds__(256) void gemm_bf16_wmma(
    const __bf16* __restrict__ A,   // [M][KK]
    const __bf16* __restrict__ B,   // [NN][KK]
    float* __restrict__ C,          // [M][NN]
    int M)
{
    __shared__ __bf16 As[BM * LDT];
    __shared__ __bf16 Bs[BN * LDT];

    const int tid  = threadIdx.x;
    const int wave = tid >> 5;
    const int lane = tid & 31;
    const int rowBase = blockIdx.y * BM;
    const int colBase = blockIdx.x * BN;
    const int wm = wave & 1;    // 64-row band
    const int wn = wave >> 1;   // 32-col band

    const v8f vzero = {0.f,0.f,0.f,0.f,0.f,0.f,0.f,0.f};
    v8f acc[4][2];
#pragma unroll
    for (int i = 0; i < 4; ++i)
#pragma unroll
        for (int j = 0; j < 2; ++j) acc[i][j] = vzero;

    const int half = lane >> 4;   // lane group 0/1
    const int m16  = lane & 15;

    // cooperative staging: each thread moves 8 bf16 (16B) per 64-row pass
    const int lrow = tid >> 2;          // 0..63
    const int lcol = (tid & 3) * 8;     // 0,8,16,24

    const __bf16* aPtr = A + (size_t)(rowBase + lrow) * KK + lcol;
    const __bf16* bPtr = B + (size_t)(colBase + lrow) * KK + lcol;

    // prefetch K-tile 0 into registers
    v8bf pa0 = *(const v8bf*)(aPtr);
    v8bf pa1 = *(const v8bf*)(aPtr + (size_t)64 * KK);
    v8bf pb0 = *(const v8bf*)(bPtr);
    v8bf pb1 = *(const v8bf*)(bPtr + (size_t)64 * KK);

    for (int k0 = 0; k0 < KK; k0 += BK) {
        __syncthreads();   // previous iteration's fragment loads are done
        *(v8bf*)&As[lrow * LDT + lcol]        = pa0;
        *(v8bf*)&As[(64 + lrow) * LDT + lcol] = pa1;
        *(v8bf*)&Bs[lrow * LDT + lcol]        = pb0;
        *(v8bf*)&Bs[(64 + lrow) * LDT + lcol] = pb1;
        __syncthreads();

        // prefetch next K-tile while this tile's WMMAs run
        if (k0 + BK < KK) {
            pa0 = *(const v8bf*)(aPtr + k0 + BK);
            pa1 = *(const v8bf*)(aPtr + (size_t)64 * KK + k0 + BK);
            pb0 = *(const v8bf*)(bPtr + k0 + BK);
            pb1 = *(const v8bf*)(bPtr + (size_t)64 * KK + k0 + BK);
        }

        // A frag (ISA 16-bit A 16x32): lane-half holds K {b..b+7} U {b+16..b+23}, b=half*8
        v16bf afrag[4];
#pragma unroll
        for (int i = 0; i < 4; ++i) {
            const __bf16* rp = &As[(wm * 64 + i * 16 + m16) * LDT];
            afrag[i] = cat16(*(const v8bf*)(rp + half * 8),
                             *(const v8bf*)(rp + half * 8 + 16));
        }
        // B frag (ISA B KxN table): lane-half holds K {b..b+7} U {b+16..b+23}, b=half*8
        v16bf bfrag[2];
#pragma unroll
        for (int j = 0; j < 2; ++j) {
            const __bf16* rp = &Bs[(wn * 32 + j * 16 + m16) * LDT];
            bfrag[j] = cat16(*(const v8bf*)(rp + half * 8),
                             *(const v8bf*)(rp + half * 8 + 16));
        }
#pragma unroll
        for (int i = 0; i < 4; ++i)
#pragma unroll
            for (int j = 0; j < 2; ++j)
                acc[i][j] = __builtin_amdgcn_wmma_f32_16x16x32_bf16(
                    false, afrag[i], false, bfrag[j],
                    (short)0, acc[i][j], false, false);
    }

    // D layout: VGPR r -> M = r + 8*half, N = lane%16.
    // NN is compile-time -> one base address per fragment + immediate offsets.
    const int nOut = colBase + wn * 32 + m16;
#pragma unroll
    for (int i = 0; i < 4; ++i) {
        const int mBase = rowBase + wm * 64 + i * 16 + half * 8;
#pragma unroll
        for (int j = 0; j < 2; ++j) {
            float* cp = C + (size_t)mBase * NN + nOut + j * 16;
#pragma unroll
            for (int r = 0; r < 8; ++r)
                cp[(size_t)r * NN] = acc[i][j][r];
        }
    }
}

// ---------------------------------------------------------------------------
// Final 512->1 linear + bias + NonSpikingIF cumsum. One wave per batch row n.
// ---------------------------------------------------------------------------
__global__ __launch_bounds__(256) void out_kernel(
    const float* __restrict__ s3,    // (T,N,F)
    const float* __restrict__ Wout,  // (1,F)
    const float* __restrict__ bout,  // (1,)
    float* __restrict__ out,         // (T,N,1)
    int T, int Nb, int F)
{
    const int gtid = blockIdx.x * blockDim.x + threadIdx.x;
    const int lane = gtid & 31;
    const int n = gtid >> 5;
    if (n >= Nb) return;  // whole waves exit together
    const float b = bout[0];
    float csum = 0.f;
    for (int t = 0; t < T; ++t) {
        float d = 0.f;
        const float* row = &s3[((size_t)t * Nb + n) * F];
        for (int j = lane; j < F; j += 32) d += row[j] * Wout[j];
        d += __shfl_xor(d, 16, 32);
        d += __shfl_xor(d, 8, 32);
        d += __shfl_xor(d, 4, 32);
        d += __shfl_xor(d, 2, 32);
        d += __shfl_xor(d, 1, 32);
        csum += d + b;
        if (lane == 0) out[(size_t)t * Nb + n] = csum;
    }
}

// ---------------------------------------------------------------------------
extern "C" void kernel_launch(void* const* d_in, const int* in_sizes, int n_in,
                              void* d_out, int out_size, void* d_ws, size_t ws_size,
                              hipStream_t stream) {
    const float* x      = (const float*)d_in[0];
    const float* w_jeff = (const float*)d_in[1];
    const float* w_cc   = (const float*)d_in[2];
    const float* w_sf0  = (const float*)d_in[3];
    const float* W1     = (const float*)d_in[4];
    const float* w_sf1  = (const float*)d_in[5];
    const float* W2     = (const float*)d_in[6];
    const float* w_sf2  = (const float*)d_in[7];
    const float* W3     = (const float*)d_in[8];
    const float* w_sf3  = (const float*)d_in[9];
    const float* W_out  = (const float*)d_in[10];
    const float* b_out  = (const float*)d_in[11];
    float* out = (float*)d_out;

    const int T = 32, Nb = 64;
    constexpr int C = 512, F1 = 2048, F2 = 1024, F3 = 512;
    const int M = T * Nb;  // 2048 GEMM rows

    // Workspace layout (31 MB total):
    //   bufA : 16 MB f32   currents / spikes ping buffer (max T*N*F1)
    //   bufH :  8 MB bf16  filtered activations (WMMA A)
    //   W*b  :  7 MB bf16  converted weights
    char* ws = (char*)d_ws;
    float*  bufA = (float*)ws;
    __bf16* bufH = (__bf16*)(ws + (size_t)16 * 1024 * 1024);
    __bf16* W1b  = (__bf16*)(ws + (size_t)24 * 1024 * 1024);
    __bf16* W2b  = (__bf16*)(ws + (size_t)26 * 1024 * 1024);
    __bf16* W3b  = (__bf16*)(ws + (size_t)30 * 1024 * 1024);

    // --- weights -> bf16 ---
    f2bf_kernel<<<1024, 256, 0, stream>>>(W1, W1b, F1 * C);
    f2bf_kernel<<<1024, 256, 0, stream>>>(W2, W2b, F2 * F1);
    f2bf_kernel<<<1024, 256, 0, stream>>>(W3, W3b, F3 * F2);

    // --- layer 0: fully fused Jeffress front-end -> spikes s0 in bufA ---
    {
        const int threads = Nb * C * 32;   // one wave per (n,c)
        l0_kernel<<<threads / 256, 256, 0, stream>>>(x, w_jeff, w_cc, w_sf0,
                                                     bufA, T, Nb, C);
    }

    // --- block 1: 512 -> 2048 ---
    {
        int NC = Nb * C;
        filter_kernel<<<(NC + 255) / 256, 256, 0, stream>>>(bufA, w_sf1, bufH, T, NC);
        dim3 g(F1 / BN, M / BM);
        gemm_bf16_wmma<F1, C><<<g, 256, 0, stream>>>(bufH, W1b, bufA, M);
        int NC1 = Nb * F1;
        if_kernel<<<(NC1 + 255) / 256, 256, 0, stream>>>(bufA, T, NC1);
    }
    // --- block 2: 2048 -> 1024 ---
    {
        int NC = Nb * F1;
        filter_kernel<<<(NC + 255) / 256, 256, 0, stream>>>(bufA, w_sf2, bufH, T, NC);
        dim3 g(F2 / BN, M / BM);
        gemm_bf16_wmma<F2, F1><<<g, 256, 0, stream>>>(bufH, W2b, bufA, M);
        int NC2 = Nb * F2;
        if_kernel<<<(NC2 + 255) / 256, 256, 0, stream>>>(bufA, T, NC2);
    }
    // --- block 3: 1024 -> 512 ---
    {
        int NC = Nb * F2;
        filter_kernel<<<(NC + 255) / 256, 256, 0, stream>>>(bufA, w_sf3, bufH, T, NC);
        dim3 g(F3 / BN, M / BM);
        gemm_bf16_wmma<F3, F2><<<g, 256, 0, stream>>>(bufH, W3b, bufA, M);
        int NC3 = Nb * F3;
        if_kernel<<<(NC3 + 255) / 256, 256, 0, stream>>>(bufA, T, NC3);
    }

    // --- head: 512 -> 1 linear + bias + cumsum over T ---
    out_kernel<<<(Nb * 32) / 256, 256, 0, stream>>>(bufA, W_out, b_out, out, T, Nb, F3);
}
